// MRASelfAttention_18399639896661
// MI455X (gfx1250) — matrix operand order repaired and checked by reading, not compiled, not run on hardware
//
#include <hip/hip_runtime.h>
#include <cstdint>
#include <cstddef>

// ---------------------------------------------------------------------------
// MRA self-attention for MI455X (gfx1250, wave32, WMMA).
// QKV GEMM and high-res block matmuls use v_wmma_f32_16x16x32_bf16; low-res ones are scalar.
// Workspace budget ~115 MB in d_ws.
// ---------------------------------------------------------------------------

typedef __attribute__((ext_vector_type(8)))  __bf16 v8bf;
typedef __attribute__((ext_vector_type(16))) __bf16 v16bf;
typedef __attribute__((ext_vector_type(8)))  float  v8f;

#define SEQ     4096
#define HDIM    1024
#define NHEADS  16
#define DH      64
#define BATCH   2
#define MB      (BATCH * NHEADS)   // 32
#define MROWS   (BATCH * SEQ)      // 8192
#define NB      128                // seq blocks
#define BS      32                 // block size
#define NSEL    512                // NUM_BLOCKS
#define SCALE   0.125f             // 1/sqrt(64)

// ---------------- WMMA fragment helpers (layouts per CDNA5 ISA 7.12.2) -----

// A 16x32 bf16 fragment: lane(<16) row=lane holds K[0..7],K[16..23];
// lane(>=16) row=lane-16 holds K[8..15],K[24..31]. p = row base + koffA.
static __device__ __forceinline__ v16bf load_fragA(const __bf16* p) {
  v8bf lo = *(const v8bf*)(p);
  v8bf hi = *(const v8bf*)(p + 16);
  return __builtin_shufflevector(lo, hi, 0,1,2,3,4,5,6,7,8,9,10,11,12,13,14,15);
}
// B 32x16 bf16 fragment: lane(<16) col=lane holds K[0..7],K[16..23];
// lane(>=16) col=lane-16 holds K[8..15],K[24..31]. p = column-source row base + kbB.
static __device__ __forceinline__ v16bf load_fragB(const __bf16* p) {
  v8bf lo = *(const v8bf*)(p);
  v8bf hi = *(const v8bf*)(p + 16);
  return __builtin_shufflevector(lo, hi, 0,1,2,3,4,5,6,7,8,9,10,11,12,13,14,15);
}
static __device__ __forceinline__ v8f wmma_bf16(v16bf a, v16bf b, v8f c) {
  return __builtin_amdgcn_wmma_f32_16x16x32_bf16(false, a, false, b, (short)0, c,
                                                 false, false);
}

static __device__ __forceinline__ void atomicMaxF(float* addr, float v) {
  if (v >= 0.f) atomicMax((int*)addr, __float_as_int(v));
  else          atomicMin((unsigned int*)addr, (unsigned int)__float_as_int(v));
}
// order-preserving float->uint map (for exact top-k threshold search)
static __device__ __forceinline__ unsigned orderKey(float f) {
  unsigned u = __float_as_uint(f);
  return (u & 0x80000000u) ? ~u : (u | 0x80000000u);
}

// ---------------- K0: fp32 -> bf16 convert ---------------------------------
__global__ void cvt_bf16_kernel(const float* __restrict__ src,
                                __bf16* __restrict__ dst, int n) {
  int i = blockIdx.x * blockDim.x + threadIdx.x;
  if (i < n) dst[i] = (__bf16)src[i];
}

// ---------------- K1: QKV projection GEMM (WMMA, LDS tiled) ----------------
// C[m, n] = sum_k hidden[m, k] * W[n, k] + bias[n], then *mask, head-split.
// WG: 256 threads = 8 waves; WG tile 128x128, wave tile 32x64, K-step 32.
__global__ __launch_bounds__(256) void qkv_gemm_kernel(
    const __bf16* __restrict__ Abf,    // 8192 x 1024
    const __bf16* __restrict__ WbfAll, // 3 x 1024 x 1024
    const float* __restrict__ b_q, const float* __restrict__ b_k,
    const float* __restrict__ b_v,
    const float* __restrict__ am,      // (2, 4096) additive mask
    __bf16* __restrict__ Qbf, __bf16* __restrict__ Kbf, __bf16* __restrict__ Vbf)
{
  const int proj = blockIdx.z;
  const __bf16* Wbf = WbfAll + (size_t)proj * HDIM * HDIM;
  const float* bias = (proj == 0) ? b_q : (proj == 1) ? b_k : b_v;
  __bf16* Dst = (proj == 0) ? Qbf : (proj == 1) ? Kbf : Vbf;

  __shared__ __align__(16) __bf16 As[128][32];
  __shared__ __align__(16) __bf16 Bs[128][32];

  const int m0 = blockIdx.y * 128;
  const int n0 = blockIdx.x * 128;
  const int tid  = threadIdx.x;
  const int lane = tid & 31;
  const int wave = tid >> 5;
  const int wm = (wave & 3) * 32;   // wave row offset inside WG tile
  const int wn = (wave >> 2) * 64;  // wave col offset inside WG tile

  const int rA    = lane & 15;
  const int koffA = (lane >> 4) * 8;
  const int kbB   = (lane >> 4) * 8;

  v8f acc[2][4];
  #pragma unroll
  for (int i = 0; i < 2; ++i)
    #pragma unroll
    for (int j = 0; j < 4; ++j)
      acc[i][j] = v8f{0.f,0.f,0.f,0.f,0.f,0.f,0.f,0.f};

  for (int k0 = 0; k0 < HDIM; k0 += 32) {
    // cooperative LDS fill: 128x32 bf16 each, 16B per access
    #pragma unroll
    for (int c = 0; c < 2; ++c) {
      int idx  = tid + c * 256;        // 512 chunks of 8 bf16
      int row  = idx >> 2;
      int col8 = (idx & 3) * 8;
      *(v8bf*)&As[row][col8] =
          *(const v8bf*)&Abf[(size_t)(m0 + row) * HDIM + k0 + col8];
      *(v8bf*)&Bs[row][col8] =
          *(const v8bf*)&Wbf[(size_t)(n0 + row) * HDIM + k0 + col8];
      if (k0 + 32 < HDIM) {  // prefetch next K tile (global_prefetch_b8)
        __builtin_prefetch(&Abf[(size_t)(m0 + row) * HDIM + k0 + 32 + col8], 0, 3);
        __builtin_prefetch(&Wbf[(size_t)(n0 + row) * HDIM + k0 + 32 + col8], 0, 3);
      }
    }
    __syncthreads();

    v16bf aF[2], bF[4];
    #pragma unroll
    for (int i = 0; i < 2; ++i)
      aF[i] = load_fragA(&As[wm + i * 16 + rA][koffA]);
    #pragma unroll
    for (int j = 0; j < 4; ++j)
      bF[j] = load_fragB(&Bs[wn + j * 16 + rA][kbB]);

    #pragma unroll
    for (int i = 0; i < 2; ++i)
      #pragma unroll
      for (int j = 0; j < 4; ++j)
        acc[i][j] = wmma_bf16(aF[i], bF[j], acc[i][j]);
    __syncthreads();
  }

  // epilogue: +bias, *mask, head-split bf16 store.
  // C layout: col n = lane&15; rows m = (lane>=16 ? 8 : 0) + r.
  const int colL  = lane & 15;
  const int rbase = (lane >> 4) * 8;
  #pragma unroll
  for (int i = 0; i < 2; ++i) {
    #pragma unroll
    for (int j = 0; j < 4; ++j) {
      int gn = n0 + wn + j * 16 + colL;
      float bv = bias[gn];
      int h = gn >> 6, dh = gn & 63;
      #pragma unroll
      for (int r = 0; r < 8; ++r) {
        int gm   = m0 + wm + i * 16 + rbase + r;
        int bb   = gm >> 12;          // / 4096
        int sRow = gm & (SEQ - 1);
        float maskv = 1.f + am[bb * SEQ + sRow] * 1e-4f;
        float val = (acc[i][j][r] + bv) * maskv;
        Dst[((size_t)(bb * NHEADS + h) * SEQ + sRow) * DH + dh] = (__bf16)val;
      }
    }
  }
}

// ---------------- K2: block means Q_hat/K_hat/V_hat + tc -------------------
__global__ __launch_bounds__(64) void block_means_kernel(
    const __bf16* __restrict__ Qbf, const __bf16* __restrict__ Kbf,
    const __bf16* __restrict__ Vbf, const float* __restrict__ am,
    float* __restrict__ Qhat, float* __restrict__ Khat, float* __restrict__ Vhat,
    float* __restrict__ tc)
{
  const int blk = blockIdx.x, mb = blockIdx.y, d = threadIdx.x;
  const int batch = mb >> 4;
  const size_t base = ((size_t)mb * SEQ + blk * BS) * DH;
  float sq = 0.f, sk = 0.f, sv = 0.f, t = 0.f;
  for (int r = 0; r < BS; ++r) {
    sq += (float)Qbf[base + r * DH + d];
    sk += (float)Kbf[base + r * DH + d];
    sv += (float)Vbf[base + r * DH + d];
    t  += 1.f + am[batch * SEQ + blk * BS + r] * 1e-4f;
  }
  float inv = 1.f / (t + 1e-6f);
  size_t o = ((size_t)mb * NB + blk) * DH + d;
  Qhat[o] = sq * inv; Khat[o] = sk * inv; Vhat[o] = sv * inv;
  if (d == 0) tc[mb * NB + blk] = t;
}

// ---------------- K3: low-res logits, row max, centered L ------------------
// L[i,j] = (Qhat_i . Khat_j)*scale - 10000*(tc_i*tc_j<0.5) - rowmax_i
__global__ __launch_bounds__(128) void low_logit_kernel(
    const float* __restrict__ Qhat, const float* __restrict__ Khat,
    const float* __restrict__ tc,
    float* __restrict__ L, float* __restrict__ rowMax)
{
  __shared__ float qrow[DH];
  __shared__ float red[NB];
  const int i = blockIdx.x, mb = blockIdx.y, j = threadIdx.x;
  if (j < DH) qrow[j] = Qhat[((size_t)mb * NB + i) * DH + j];
  __syncthreads();
  const float* Kj = Khat + ((size_t)mb * NB + j) * DH;
  float s = 0.f;
  for (int d = 0; d < DH; ++d) s += qrow[d] * Kj[d];
  s *= SCALE;
  red[j] = s;                              // row max BEFORE mask subtraction
  __syncthreads();
  for (int off = 64; off > 0; off >>= 1) {
    if (j < off) red[j] = fmaxf(red[j], red[j + off]);
    __syncthreads();
  }
  float rm = red[0];
  float tci = tc[mb * NB + i], tcj = tc[mb * NB + j];
  float sm = s - (((tci * tcj) < 0.5f) ? 10000.f : 0.f);
  L[((size_t)mb * NB + i) * NB + j] = sm - rm;
  if (j == 0) rowMax[mb * NB + i] = rm;
}

// ---------------- K4: init scratch ----------------------------------------
__global__ void init_kernel(float* __restrict__ maxBlocks, float* __restrict__ hiOut,
                            float* __restrict__ hiNorm, unsigned* __restrict__ selCount)
{
  size_t i = (size_t)blockIdx.x * 256 + threadIdx.x; // grid == MB*SEQ*DH
  hiOut[i] = 0.f;
  if (i < (size_t)MB * SEQ) { maxBlocks[i] = -1e8f; hiNorm[i] = 0.f; }
  if (i < MB) selCount[i] = 0u;
}

// ---------------- K5: exact top-512 threshold + index collection -----------
__global__ __launch_bounds__(256) void select_kernel(
    const float* __restrict__ L, unsigned* __restrict__ thrU,
    unsigned* __restrict__ sel, unsigned* __restrict__ selCount)
{
  const int mb = blockIdx.x, tid = threadIdx.x;
  const float* Lm = L + (size_t)mb * NB * NB;
  __shared__ unsigned cnt[256];
  __shared__ unsigned s_lo, s_hi;
  if (tid == 0) { s_lo = 0u; s_hi = 0xFFFFFFFFu; }
  __syncthreads();
  // lrn = L + 5000*band(|i-j|<=1); find max t with count(key>=t) >= 512
  for (;;) {
    __syncthreads();
    unsigned lo = s_lo, hi = s_hi;
    if (lo >= hi) break;
    unsigned mid = lo + ((hi - lo) >> 1) + 1u;
    unsigned c = 0;
    for (int idx = tid; idx < NB * NB; idx += 256) {
      int i = idx >> 7, j = idx & 127;
      float v = Lm[idx];
      int di = i - j;
      if (di >= -1 && di <= 1) v += 5000.f;
      c += (orderKey(v) >= mid) ? 1u : 0u;
    }
    cnt[tid] = c;
    __syncthreads();
    for (int s2 = 128; s2 > 0; s2 >>= 1) {
      if (tid < s2) cnt[tid] += cnt[tid + s2];
      __syncthreads();
    }
    if (tid == 0) { if (cnt[0] >= NSEL) s_lo = mid; else s_hi = mid - 1u; }
  }
  __syncthreads();
  unsigned T = s_lo;
  if (tid == 0) thrU[mb] = T;
  for (int idx = tid; idx < NB * NB; idx += 256) {
    int i = idx >> 7, j = idx & 127;
    float v = Lm[idx];
    int di = i - j;
    if (di >= -1 && di <= 1) v += 5000.f;
    if (orderKey(v) >= T) {
      unsigned pos = atomicAdd(&selCount[mb], 1u);
      if (pos < NSEL) sel[mb * NSEL + pos] = ((unsigned)i << 8) | (unsigned)j;
    }
  }
}

// ---------------- K6: high-res pass A: per-query-row block max -------------
// One wave per selected (q_blk,k_blk) pair; 32x32x64 logits via 8 WMMAs.
__global__ __launch_bounds__(32) void hi_max_kernel(
    const __bf16* __restrict__ Qbf, const __bf16* __restrict__ Kbf,
    const unsigned* __restrict__ sel, float* __restrict__ maxBlocks)
{
  const int mb = blockIdx.y;
  const unsigned pc = sel[mb * NSEL + blockIdx.x];
  const int qb = (int)(pc >> 8), kb = (int)(pc & 255u);
  const __bf16* Ksp = Kbf + ((size_t)mb * SEQ + kb * BS) * DH;
  const __bf16* Qsp = Qbf + ((size_t)mb * SEQ + qb * BS) * DH;
  const int lane = threadIdx.x;
  const int rA = lane & 15, koffA = (lane >> 4) * 8, kbB = (lane >> 4) * 8;

  float cmax[2] = {-3.0e38f, -3.0e38f};
  #pragma unroll
  for (int qt = 0; qt < 2; ++qt) {
    #pragma unroll
    for (int kt = 0; kt < 2; ++kt) {
      v8f c = v8f{0.f,0.f,0.f,0.f,0.f,0.f,0.f,0.f};
      #pragma unroll
      for (int kk = 0; kk < DH; kk += 32) {
        v16bf aF = load_fragA(Ksp + (kt * 16 + rA) * DH + kk + koffA);
        v16bf bF = load_fragB(Qsp + (qt * 16 + rA) * DH + kk + kbB);
        c = wmma_bf16(aF, bF, c);
      }
      float m = c[0];
      #pragma unroll
      for (int r = 1; r < 8; ++r) m = fmaxf(m, c[r]);
      cmax[qt] = fmaxf(cmax[qt], m);
    }
  }
  cmax[0] = fmaxf(cmax[0], __shfl_xor(cmax[0], 16, 32));
  cmax[1] = fmaxf(cmax[1], __shfl_xor(cmax[1], 16, 32));
  if (lane < 16) {
    atomicMaxF(&maxBlocks[mb * SEQ + qb * BS + lane],      cmax[0] * SCALE);
    atomicMaxF(&maxBlocks[mb * SEQ + qb * BS + 16 + lane], cmax[1] * SCALE);
  }
}

// ---------------- K7: high-res pass B: attn, attn^T @ V, scatter -----------
__global__ __launch_bounds__(32) void hi_acc_kernel(
    const __bf16* __restrict__ Qbf, const __bf16* __restrict__ Kbf,
    const __bf16* __restrict__ Vbf, const unsigned* __restrict__ sel,
    const float* __restrict__ maxBlocks, const float* __restrict__ am,
    float* __restrict__ hiOut, float* __restrict__ hiNorm)
{
  __shared__ __align__(16) __bf16 attnT[BS][BS];   // [q][k]
  __shared__ __align__(16) __bf16 VsT[DH][BS];     // [d][k]
  const int mb = blockIdx.y;
  const unsigned pc = sel[mb * NSEL + blockIdx.x];
  const int qb = (int)(pc >> 8), kbId = (int)(pc & 255u);
  const __bf16* Ksp = Kbf + ((size_t)mb * SEQ + kbId * BS) * DH;
  const __bf16* Qsp = Qbf + ((size_t)mb * SEQ + qb * BS) * DH;
  const __bf16* Vsp = Vbf + ((size_t)mb * SEQ + kbId * BS) * DH;
  const int lane = threadIdx.x;
  const int rA = lane & 15, koffA = (lane >> 4) * 8, kbB = (lane >> 4) * 8;
  const int rbase = (lane >> 4) * 8;
  const int batch = mb >> 4;
  const int qBase = qb * BS;

  // stage V^T into LDS: VsT[d][k]
  for (int d = 0; d < DH; ++d) VsT[d][lane] = Vsp[lane * DH + d];

  float nrm[2] = {0.f, 0.f};
  #pragma unroll
  for (int qt = 0; qt < 2; ++qt) {
    const float mx = maxBlocks[mb * SEQ + qBase + qt * 16 + (lane & 15)];
    #pragma unroll
    for (int kt = 0; kt < 2; ++kt) {
      v8f c = v8f{0.f,0.f,0.f,0.f,0.f,0.f,0.f,0.f};
      #pragma unroll
      for (int kk = 0; kk < DH; kk += 32) {
        v16bf aF = load_fragA(Ksp + (kt * 16 + rA) * DH + kk + koffA);
        v16bf bF = load_fragB(Qsp + (qt * 16 + rA) * DH + kk + kbB);
        c = wmma_bf16(aF, bF, c);
      }
      v8bf pk;
      #pragma unroll
      for (int r = 0; r < 8; ++r) {
        int krow = kt * 16 + rbase + r;
        float mk = 1.f + am[batch * SEQ + kbId * BS + krow] * 1e-4f;
        float a = __expf(c[r] * SCALE - mx - 10000.f * (1.f - mk));
        nrm[qt] += a;
        pk[r] = (__bf16)a;
      }
      // attn^T: lane holds column q=(lane&15) of C; write rows of attnT[q][k]
      *(v8bf*)&attnT[qt * 16 + (lane & 15)][kt * 16 + rbase] = pk;
    }
  }
  __syncthreads();

  // out[q][d] = attnT(32x32) @ Vs(32x64), single K-step of 32
  v16bf aQ[2];
  #pragma unroll
  for (int qt = 0; qt < 2; ++qt) aQ[qt] = load_fragA(&attnT[qt * 16 + rA][koffA]);
  #pragma unroll
  for (int dt = 0; dt < 4; ++dt) {
    v16bf bV = load_fragB(&VsT[dt * 16 + rA][kbB]);
    #pragma unroll
    for (int qt = 0; qt < 2; ++qt) {
      v8f o = wmma_bf16(aQ[qt], bV, v8f{0.f,0.f,0.f,0.f,0.f,0.f,0.f,0.f});
      int dcol = dt * 16 + (lane & 15);
      #pragma unroll
      for (int r = 0; r < 8; ++r) {
        int qrow = qt * 16 + rbase + r;
        atomicAdd(&hiOut[((size_t)mb * SEQ + qBase + qrow) * DH + dcol], o[r]);
      }
    }
  }
  nrm[0] += __shfl_xor(nrm[0], 16, 32);
  nrm[1] += __shfl_xor(nrm[1], 16, 32);
  if (lane < 16) {
    atomicAdd(&hiNorm[mb * SEQ + qBase + lane],      nrm[0]);
    atomicAdd(&hiNorm[mb * SEQ + qBase + 16 + lane], nrm[1]);
  }
}

// ---------------- K8: low-res branch (per query block) ---------------------
__global__ __launch_bounds__(128) void low_out_kernel(
    const float* __restrict__ L, const float* __restrict__ tc,
    const float* __restrict__ Vhat, const unsigned* __restrict__ thrU,
    float* __restrict__ lowOut, float* __restrict__ lowNorm)
{
  __shared__ float attn[NB];
  const int i = blockIdx.x, mb = blockIdx.y, j = threadIdx.x;
  float Lij = L[((size_t)mb * NB + i) * NB + j];
  int di = i - j;
  float lrn = Lij + ((di >= -1 && di <= 1) ? 5000.f : 0.f);
  float hm = (orderKey(lrn) >= thrU[mb]) ? 1.f : 0.f;
  float a = __expf(Lij - 10000.f * hm) * tc[mb * NB + j];
  attn[j] = a;
  __syncthreads();
  if (j < DH) {
    float s = 0.f;
    for (int jj = 0; jj < NB; ++jj)
      s += attn[jj] * Vhat[((size_t)mb * NB + jj) * DH + j];
    lowOut[((size_t)mb * NB + i) * DH + j] = s;
  } else if (j == DH) {
    float s = 0.f;
    for (int jj = 0; jj < NB; ++jj) s += attn[jj];
    lowNorm[mb * NB + i] = s;
  }
}

// ---------------- K9: merge branches, un-split heads -----------------------
__global__ __launch_bounds__(256) void finalize_kernel(
    const float* __restrict__ hiOut, const float* __restrict__ hiNorm,
    const float* __restrict__ lowOut, const float* __restrict__ lowNorm,
    const float* __restrict__ rowMax, const float* __restrict__ maxBlocks,
    const float* __restrict__ am, float* __restrict__ out)
{
  size_t idx = (size_t)blockIdx.x * 256 + threadIdx.x;  // grid == 2*4096*1024
  int b = (int)(idx >> 22);
  int rem = (int)(idx & ((1u << 22) - 1));
  int s = rem >> 10;
  int j = rem & 1023;
  int h = j >> 6, dh = j & 63;
  int mb = b * NHEADS + h;
  int i = s >> 5;
  float mask = 1.f + am[b * SEQ + s] * 1e-4f;
  float logc = (rowMax[mb * NB + i] - maxBlocks[mb * SEQ + s]) * mask;
  float lowc = __expf(fminf(logc, 0.f));
  float hic  = __expf(-fmaxf(logc, 0.f));
  float ho = hiOut[((size_t)mb * SEQ + s) * DH + dh];
  float lo = lowOut[((size_t)mb * NB + i) * DH + dh];
  float hn = hiNorm[mb * SEQ + s];
  float ln = lowNorm[mb * NB + i];
  float ctx = (ho * hic + lo * lowc) / (hn * hic + ln * lowc + 1e-6f);
  out[idx] = ctx * mask;
}

// ---------------------------------------------------------------------------
extern "C" void kernel_launch(void* const* d_in, const int* in_sizes, int n_in,
                              void* d_out, int out_size, void* d_ws, size_t ws_size,
                              hipStream_t stream)
{
  const float* hidden = (const float*)d_in[0];
  const float* am     = (const float*)d_in[1];
  const float* w_q = (const float*)d_in[2];
  const float* b_q = (const float*)d_in[3];
  const float* w_k = (const float*)d_in[4];
  const float* b_k = (const float*)d_in[5];
  const float* w_v = (const float*)d_in[6];
  const float* b_v = (const float*)d_in[7];
  float* out = (float*)d_out;

  char* ws = (char*)d_ws;
  size_t off = 0;
  auto alloc = [&](size_t bytes) -> void* {
    void* p = ws + off;
    off += (bytes + 255) & ~(size_t)255;
    return p;
  };

  __bf16* hidBf = (__bf16*)alloc((size_t)MROWS * HDIM * 2);
  __bf16* wBf   = (__bf16*)alloc((size_t)3 * HDIM * HDIM * 2);
  __bf16* Qbf   = (__bf16*)alloc((size_t)MB * SEQ * DH * 2);
  __bf16* Kbf   = (__bf16*)alloc((size_t)MB * SEQ * DH * 2);
  __bf16* Vbf   = (__bf16*)alloc((size_t)MB * SEQ * DH * 2);
  float* Qhat = (float*)alloc((size_t)MB * NB * DH * 4);
  float* Khat = (float*)alloc((size_t)MB * NB * DH * 4);
  float* Vhat = (float*)alloc((size_t)MB * NB * DH * 4);
  float* tc   = (float*)alloc((size_t)MB * NB * 4);
  float* L    = (float*)alloc((size_t)MB * NB * NB * 4);
  float* rowMax = (float*)alloc((size_t)MB * NB * 4);
  unsigned* thrU     = (unsigned*)alloc(MB * 4);
  unsigned* selCount = (unsigned*)alloc(MB * 4);
  unsigned* sel      = (unsigned*)alloc((size_t)MB * NSEL * 4);
  float* maxBlocks = (float*)alloc((size_t)MB * SEQ * 4);
  float* hiOut     = (float*)alloc((size_t)MB * SEQ * DH * 4);
  float* hiNorm    = (float*)alloc((size_t)MB * SEQ * 4);
  float* lowOut    = (float*)alloc((size_t)MB * NB * DH * 4);
  float* lowNorm   = (float*)alloc((size_t)MB * NB * 4);
  (void)ws_size; (void)in_sizes; (void)n_in; (void)out_size;

  const int nHid = MROWS * HDIM;
  const int nW   = HDIM * HDIM;
  cvt_bf16_kernel<<<(nHid + 255) / 256, 256, 0, stream>>>(hidden, hidBf, nHid);
  cvt_bf16_kernel<<<(nW + 255) / 256, 256, 0, stream>>>(w_q, wBf,          nW);
  cvt_bf16_kernel<<<(nW + 255) / 256, 256, 0, stream>>>(w_k, wBf + nW,     nW);
  cvt_bf16_kernel<<<(nW + 255) / 256, 256, 0, stream>>>(w_v, wBf + 2 * nW, nW);

  qkv_gemm_kernel<<<dim3(HDIM / 128, MROWS / 128, 3), 256, 0, stream>>>(
      hidBf, wBf, b_q, b_k, b_v, am, Qbf, Kbf, Vbf);

  block_means_kernel<<<dim3(NB, MB), 64, 0, stream>>>(
      Qbf, Kbf, Vbf, am, Qhat, Khat, Vhat, tc);

  low_logit_kernel<<<dim3(NB, MB), NB, 0, stream>>>(Qhat, Khat, tc, L, rowMax);

  init_kernel<<<(MB * SEQ * DH) / 256, 256, 0, stream>>>(
      maxBlocks, hiOut, hiNorm, selCount);

  select_kernel<<<MB, 256, 0, stream>>>(L, thrU, sel, selCount);

  hi_max_kernel<<<dim3(NSEL, MB), 32, 0, stream>>>(Qbf, Kbf, sel, maxBlocks);

  hi_acc_kernel<<<dim3(NSEL, MB), 32, 0, stream>>>(
      Qbf, Kbf, Vbf, sel, maxBlocks, am, hiOut, hiNorm);

  low_out_kernel<<<dim3(NB, MB), NB, 0, stream>>>(
      L, tc, Vhat, thrU, lowOut, lowNorm);

  finalize_kernel<<<(BATCH * SEQ * HDIM) / 256, 256, 0, stream>>>(
      hiOut, hiNorm, lowOut, lowNorm, rowMax, maxBlocks, am, out);
}
